// CrossAttention_65051574665735
// MI455X (gfx1250) — matrix operand order repaired and checked by reading, not compiled, not run on hardware
//
#include <hip/hip_runtime.h>
#include <hip/hip_bf16.h>

#define DEVFN __device__ __forceinline__

typedef __attribute__((ext_vector_type(16))) _Float16 v16h;
typedef __attribute__((ext_vector_type(8)))  _Float16 v8h;
typedef __attribute__((ext_vector_type(8)))  float    v8f;
typedef __attribute__((ext_vector_type(4)))  unsigned v4u;
typedef __attribute__((ext_vector_type(4)))  int      v4i;
typedef __attribute__((ext_vector_type(8)))  int      v8i;

constexpr int Bb    = 2;
constexpr int Nn    = 2048;
constexpr int Jj    = 2048;
constexpr int DIMc  = 1024;
constexpr int HEADS = 16;
constexpr int DHc   = 64;
constexpr int INNER = HEADS * DHc;   // 1024
constexpr int FFc   = 4 * DIMc;      // 4096
constexpr float EPSc = 1e-5f;

#if __has_builtin(__builtin_amdgcn_tensor_load_to_lds)
#define USE_TDM 1
#else
#define USE_TDM 0
#endif

DEVFN v8f vzero() {
  v8f z = {0.f,0.f,0.f,0.f,0.f,0.f,0.f,0.f};
  return z;
}

DEVFN v8f wmma_f16(v16h a, v16h b, v8f c) {
  return __builtin_amdgcn_wmma_f32_16x16x32_f16(
      false, a, false, b, (short)0, c, false, false);
}

// A-fragment: 16x32 f16 tile, row-major source with leading dim lda.
// lane l<16: row m0+l, K = kb+{0..7,16..23}; lane l+16: K = kb+{8..15,24..31}.
DEVFN v16h load_am(const _Float16* A, int lda, int m0, int kb, int lane) {
  const int row = m0 + (lane & 15);
  const int ko  = kb + ((lane & 16) ? 8 : 0);
  const _Float16* p = A + (size_t)row * lda + ko;
  v8h lo = *(const v8h*)p;
  v8h hi = *(const v8h*)(p + 16);
  return __builtin_shufflevector(lo, hi, 0,1,2,3,4,5,6,7,8,9,10,11,12,13,14,15);
}

// B-fragment from BT stored [N][K]: lane n<16: col n0+n, K kb+{0..7,16..23};
// lane n+16: K kb+{8..15,24..31}.
DEVFN v16h load_bt(const _Float16* BT, int ldb, int n0, int kb, int lane) {
  const int col = n0 + (lane & 15);
  const int ko  = kb + ((lane & 16) ? 8 : 0);
  const _Float16* p = BT + (size_t)col * ldb + ko;
  v8h lo = *(const v8h*)p;
  v8h hi = *(const v8h*)(p + 16);
  return __builtin_shufflevector(lo, hi, 0,1,2,3,4,5,6,7,8,9,10,11,12,13,14,15);
}

// ---------------------------------------------------------------------------
// TDM: issue a 2D tile load (tile_rows x 64 f16 elements, row stride
// row_stride_elems) from global into LDS. One wave issues; TENSORcnt tracks.
// D# layout per CDNA5 ISA ch.8 (group0 128b, group1 256b, groups2/3 zero=2D).
// ---------------------------------------------------------------------------
DEVFN void tdm_load_tile_2d(const _Float16* gsrc, _Float16* lds_dst,
                            int tile_rows, int row_stride_elems, int tensor_rows) {
#if USE_TDM
  unsigned long long ga = (unsigned long long)(size_t)gsrc;
  unsigned lds_off = (unsigned)(unsigned long long)(size_t)lds_dst;  // low 32 = LDS offset
  v4u g0;
  g0.x = 1u;                                            // count=1, user D#
  g0.y = lds_off;                                       // lds_addr
  g0.z = (unsigned)ga;                                  // global_addr[31:0]
  g0.w = (unsigned)((ga >> 32) & 0x01FFFFFFu) | (2u << 30);  // addr[56:32], type=2
  unsigned td0 = (unsigned)row_stride_elems;            // tensor_dim0 (elems)
  unsigned td1 = (unsigned)tensor_rows;                 // tensor_dim1 (rows)
  v8i g1;
  g1[0] = (int)(1u << 16);                              // data_size=1 -> 2 bytes
  g1[1] = (int)((td0 & 0xFFFFu) << 16);                 // tensor_dim0[15:0]
  g1[2] = (int)(((td0 >> 16) & 0xFFFFu) | ((td1 & 0xFFFFu) << 16));
  g1[3] = (int)(((td1 >> 16) & 0xFFFFu) | (64u << 16)); // tile_dim0 = 64 elems
  g1[4] = (int)((unsigned)tile_rows & 0xFFFFu);         // tile_dim1; tile_dim2=0
  g1[5] = (int)td0;                                     // tensor_dim0_stride[31:0]
  g1[6] = 0;
  g1[7] = 0;
  v4i z4 = {0,0,0,0};
#if __clang_major__ >= 23
  v8i z8 = {0,0,0,0,0,0,0,0};
  __builtin_amdgcn_tensor_load_to_lds(g0, g1, z4, z4, z8, 0);
#else
  __builtin_amdgcn_tensor_load_to_lds(g0, g1, z4, z4, 0);
#endif
#else
  (void)gsrc; (void)lds_dst; (void)tile_rows; (void)row_stride_elems; (void)tensor_rows;
#endif
}

// Fallback synchronous stage (rows x 64 halves), 256 threads.
DEVFN void stage_fallback(_Float16* dst, const _Float16* src, int rows,
                          int row_stride, int tid) {
  const int chunks = rows * 8;             // 8-half (16B) chunks per row
  for (int c = tid; c < chunks; c += 256) {
    int r = c >> 3, co = (c & 7) << 3;
    *(v8h*)(dst + r * 64 + co) = *(const v8h*)(src + (size_t)r * row_stride + co);
  }
}

// ---------------------------------------------------------------------------
// LDS-staged, double-buffered GEMM core.
// Block: 256 threads / 8 waves. Block tile 128(M) x 128(N), K-chunk 64.
// Wave tile 32x64 (2x4 WMMA tiles). A row-major [M][K], BT [N][K].
// ---------------------------------------------------------------------------
DEVFN void gemm_staged(const _Float16* __restrict__ A, const _Float16* __restrict__ BT,
                       int K, int Mtot, int Ntot, int bm0, int bn0,
                       _Float16* As, _Float16* Bs, int tid, v8f acc[2][4]) {
  const int lane = tid & 31;
  const int wm = ((tid >> 5) >> 1) * 32;   // 0,32,64,96
  const int wn = ((tid >> 5) & 1) * 64;    // 0,64
#pragma unroll
  for (int mi = 0; mi < 2; ++mi)
#pragma unroll
    for (int ni = 0; ni < 4; ++ni) acc[mi][ni] = vzero();

#if USE_TDM
  if (tid < 32) {
    tdm_load_tile_2d(A  + (size_t)bm0 * K, As, 128, K, Mtot);
    tdm_load_tile_2d(BT + (size_t)bn0 * K, Bs, 128, K, Ntot);
  }
#else
  stage_fallback(As, A  + (size_t)bm0 * K, 128, K, tid);
  stage_fallback(Bs, BT + (size_t)bn0 * K, 128, K, tid);
#endif

  for (int k0 = 0; k0 < K; k0 += 64) {
    const int cur = (k0 >> 6) & 1;
    _Float16* Ac = As + cur * (128 * 64);
    _Float16* Bc = Bs + cur * (128 * 64);
#if USE_TDM
    if (tid < 32) __builtin_amdgcn_s_wait_tensorcnt(0);
#endif
    __syncthreads();
    if (k0 + 64 < K) {
      _Float16* An = As + (cur ^ 1) * (128 * 64);
      _Float16* Bn = Bs + (cur ^ 1) * (128 * 64);
#if USE_TDM
      if (tid < 32) {
        tdm_load_tile_2d(A  + (size_t)bm0 * K + (k0 + 64), An, 128, K, Mtot);
        tdm_load_tile_2d(BT + (size_t)bn0 * K + (k0 + 64), Bn, 128, K, Ntot);
      }
#else
      stage_fallback(An, A  + (size_t)bm0 * K + (k0 + 64), 128, K, tid);
      stage_fallback(Bn, BT + (size_t)bn0 * K + (k0 + 64), 128, K, tid);
#endif
    }
#pragma unroll
    for (int kk = 0; kk < 64; kk += 32) {
      v16h a0 = load_am(Ac, 64, wm,      kk, lane);
      v16h a1 = load_am(Ac, 64, wm + 16, kk, lane);
      v16h b0 = load_bt(Bc, 64, wn,      kk, lane);
      v16h b1 = load_bt(Bc, 64, wn + 16, kk, lane);
      v16h b2 = load_bt(Bc, 64, wn + 32, kk, lane);
      v16h b3 = load_bt(Bc, 64, wn + 48, kk, lane);
      acc[0][0] = wmma_f16(a0, b0, acc[0][0]);
      acc[0][1] = wmma_f16(a0, b1, acc[0][1]);
      acc[0][2] = wmma_f16(a0, b2, acc[0][2]);
      acc[0][3] = wmma_f16(a0, b3, acc[0][3]);
      acc[1][0] = wmma_f16(a1, b0, acc[1][0]);
      acc[1][1] = wmma_f16(a1, b1, acc[1][1]);
      acc[1][2] = wmma_f16(a1, b2, acc[1][2]);
      acc[1][3] = wmma_f16(a1, b3, acc[1][3]);
    }
  }
}

// C layout: element v of a 16x16 f32 acc -> row tile_m + v + (lane&16?8:0),
// col tile_n + (lane&15).

// ---------- GEMM, f16 output with scale (q projection) ----------
__global__ __launch_bounds__(256) void gemm_f16out(
    const _Float16* __restrict__ A, const _Float16* __restrict__ BT,
    _Float16* __restrict__ C, int K, int M, int N, float scale) {
  __shared__ alignas(16) _Float16 As[2 * 128 * 64];
  __shared__ alignas(16) _Float16 Bs[2 * 128 * 64];
  const int tid = threadIdx.x, lane = tid & 31, wave = tid >> 5;
  const int bm0 = blockIdx.x * 128, bn0 = blockIdx.y * 128;
  v8f acc[2][4];
  gemm_staged(A, BT, K, M, N, bm0, bn0, As, Bs, tid, acc);
  const int m0 = bm0 + (wave >> 1) * 32;
  const int n0 = bn0 + (wave & 1) * 64;
#pragma unroll
  for (int mi = 0; mi < 2; ++mi)
#pragma unroll
    for (int ni = 0; ni < 4; ++ni)
#pragma unroll
      for (int v = 0; v < 8; ++v) {
        int row = m0 + mi * 16 + v + ((lane & 16) ? 8 : 0);
        int col = n0 + ni * 16 + (lane & 15);
        C[(size_t)row * N + col] = (_Float16)(acc[mi][ni][v] * scale);
      }
}

// ---------- GEMM, f32 store (attention output projection) ----------
__global__ __launch_bounds__(256) void gemm_f32store(
    const _Float16* __restrict__ A, const _Float16* __restrict__ BT,
    float* __restrict__ C, int K, int M, int N) {
  __shared__ alignas(16) _Float16 As[2 * 128 * 64];
  __shared__ alignas(16) _Float16 Bs[2 * 128 * 64];
  const int tid = threadIdx.x, lane = tid & 31, wave = tid >> 5;
  const int bm0 = blockIdx.x * 128, bn0 = blockIdx.y * 128;
  v8f acc[2][4];
  gemm_staged(A, BT, K, M, N, bm0, bn0, As, Bs, tid, acc);
  const int m0 = bm0 + (wave >> 1) * 32;
  const int n0 = bn0 + (wave & 1) * 64;
#pragma unroll
  for (int mi = 0; mi < 2; ++mi)
#pragma unroll
    for (int ni = 0; ni < 4; ++ni)
#pragma unroll
      for (int v = 0; v < 8; ++v) {
        int row = m0 + mi * 16 + v + ((lane & 16) ? 8 : 0);
        int col = n0 + ni * 16 + (lane & 15);
        C[(size_t)row * N + col] = acc[mi][ni][v];
      }
}

// ---------- GEMM, f32 accumulate-add (FF down projection) ----------
__global__ __launch_bounds__(256) void gemm_f32add(
    const _Float16* __restrict__ A, const _Float16* __restrict__ BT,
    float* __restrict__ C, int K, int M, int N) {
  __shared__ alignas(16) _Float16 As[2 * 128 * 64];
  __shared__ alignas(16) _Float16 Bs[2 * 128 * 64];
  const int tid = threadIdx.x, lane = tid & 31, wave = tid >> 5;
  const int bm0 = blockIdx.x * 128, bn0 = blockIdx.y * 128;
  v8f acc[2][4];
  gemm_staged(A, BT, K, M, N, bm0, bn0, As, Bs, tid, acc);
  const int m0 = bm0 + (wave >> 1) * 32;
  const int n0 = bn0 + (wave & 1) * 64;
#pragma unroll
  for (int mi = 0; mi < 2; ++mi)
#pragma unroll
    for (int ni = 0; ni < 4; ++ni)
#pragma unroll
      for (int v = 0; v < 8; ++v) {
        int row = m0 + mi * 16 + v + ((lane & 16) ? 8 : 0);
        int col = n0 + ni * 16 + (lane & 15);
        size_t idx = (size_t)row * N + col;
        C[idx] = C[idx] + acc[mi][ni][v];
      }
}

// ---------- kv projection: N==128; cols 0..63 -> K, 64..127 -> V^T ----------
__global__ __launch_bounds__(256) void gemm_kv(
    const _Float16* __restrict__ A, const _Float16* __restrict__ BT,
    _Float16* __restrict__ Kout, _Float16* __restrict__ VTout, int K) {
  __shared__ alignas(16) _Float16 As[2 * 128 * 64];
  __shared__ alignas(16) _Float16 Bs[2 * 128 * 64];
  const int tid = threadIdx.x, lane = tid & 31, wave = tid >> 5;
  const int bm0 = blockIdx.x * 128;
  v8f acc[2][4];
  gemm_staged(A, BT, K, Bb * Jj, 2 * DHc, bm0, 0, As, Bs, tid, acc);
  const int m0 = bm0 + (wave >> 1) * 32;
  const int n0 = (wave & 1) * 64;
#pragma unroll
  for (int mi = 0; mi < 2; ++mi)
#pragma unroll
    for (int ni = 0; ni < 4; ++ni)
#pragma unroll
      for (int v = 0; v < 8; ++v) {
        int row = m0 + mi * 16 + v + ((lane & 16) ? 8 : 0);  // b*J + j
        int col = n0 + ni * 16 + (lane & 15);
        float val = acc[mi][ni][v];
        if (col < DHc) {
          Kout[(size_t)row * DHc + col] = (_Float16)val;
        } else {
          int bidx = row >> 11;
          int j    = row & (Jj - 1);
          int d    = col - DHc;
          VTout[((size_t)bidx * DHc + d) * Jj + j] = (_Float16)val;
        }
      }
}

// ---------- gated FF1: staged A + two staged B tiles (a, gate) ----------
__global__ __launch_bounds__(256) void gemm_ff1_gated(
    const _Float16* __restrict__ A, const _Float16* __restrict__ BT,
    _Float16* __restrict__ Hout, int K) {
  __shared__ alignas(16) _Float16 As [2 * 128 * 64];
  __shared__ alignas(16) _Float16 Bsa[2 * 64 * 64];
  __shared__ alignas(16) _Float16 Bsg[2 * 64 * 64];
  const int tid = threadIdx.x, lane = tid & 31, wave = tid >> 5;
  const int bm0 = blockIdx.x * 128, bn0 = blockIdx.y * 64;
  const int wm = (wave >> 1) * 32, wn = (wave & 1) * 32;

  v8f aa[2][2], ag[2][2];
#pragma unroll
  for (int mi = 0; mi < 2; ++mi)
#pragma unroll
    for (int ni = 0; ni < 2; ++ni) { aa[mi][ni] = vzero(); ag[mi][ni] = vzero(); }

#if USE_TDM
  if (tid < 32) {
    tdm_load_tile_2d(A  + (size_t)bm0 * K,          As,  128, K, Bb * Nn);
    tdm_load_tile_2d(BT + (size_t)bn0 * K,          Bsa, 64,  K, 2 * FFc);
    tdm_load_tile_2d(BT + (size_t)(bn0 + FFc) * K,  Bsg, 64,  K, 2 * FFc);
  }
#else
  stage_fallback(As,  A  + (size_t)bm0 * K,         128, K, tid);
  stage_fallback(Bsa, BT + (size_t)bn0 * K,         64,  K, tid);
  stage_fallback(Bsg, BT + (size_t)(bn0 + FFc) * K, 64,  K, tid);
#endif

  for (int k0 = 0; k0 < K; k0 += 64) {
    const int cur = (k0 >> 6) & 1;
    _Float16* Ac  = As  + cur * (128 * 64);
    _Float16* Bac = Bsa + cur * (64 * 64);
    _Float16* Bgc = Bsg + cur * (64 * 64);
#if USE_TDM
    if (tid < 32) __builtin_amdgcn_s_wait_tensorcnt(0);
#endif
    __syncthreads();
    if (k0 + 64 < K) {
      const int nxt = cur ^ 1;
#if USE_TDM
      if (tid < 32) {
        tdm_load_tile_2d(A  + (size_t)bm0 * K + (k0 + 64),         As  + nxt * (128 * 64), 128, K, Bb * Nn);
        tdm_load_tile_2d(BT + (size_t)bn0 * K + (k0 + 64),         Bsa + nxt * (64 * 64),  64,  K, 2 * FFc);
        tdm_load_tile_2d(BT + (size_t)(bn0 + FFc) * K + (k0 + 64), Bsg + nxt * (64 * 64),  64,  K, 2 * FFc);
      }
#else
      stage_fallback(As  + nxt * (128 * 64), A  + (size_t)bm0 * K + (k0 + 64),         128, K, tid);
      stage_fallback(Bsa + nxt * (64 * 64),  BT + (size_t)bn0 * K + (k0 + 64),         64,  K, tid);
      stage_fallback(Bsg + nxt * (64 * 64),  BT + (size_t)(bn0 + FFc) * K + (k0 + 64), 64,  K, tid);
#endif
    }
#pragma unroll
    for (int kk = 0; kk < 64; kk += 32) {
      v16h a0  = load_am(Ac, 64, wm,      kk, lane);
      v16h a1  = load_am(Ac, 64, wm + 16, kk, lane);
      v16h ba0 = load_bt(Bac, 64, wn,      kk, lane);
      v16h ba1 = load_bt(Bac, 64, wn + 16, kk, lane);
      v16h bg0 = load_bt(Bgc, 64, wn,      kk, lane);
      v16h bg1 = load_bt(Bgc, 64, wn + 16, kk, lane);
      aa[0][0] = wmma_f16(a0, ba0, aa[0][0]);
      aa[0][1] = wmma_f16(a0, ba1, aa[0][1]);
      aa[1][0] = wmma_f16(a1, ba0, aa[1][0]);
      aa[1][1] = wmma_f16(a1, ba1, aa[1][1]);
      ag[0][0] = wmma_f16(a0, bg0, ag[0][0]);
      ag[0][1] = wmma_f16(a0, bg1, ag[0][1]);
      ag[1][0] = wmma_f16(a1, bg0, ag[1][0]);
      ag[1][1] = wmma_f16(a1, bg1, ag[1][1]);
    }
  }
#pragma unroll
  for (int mi = 0; mi < 2; ++mi)
#pragma unroll
    for (int ni = 0; ni < 2; ++ni)
#pragma unroll
      for (int v = 0; v < 8; ++v) {
        int row = bm0 + wm + mi * 16 + v + ((lane & 16) ? 8 : 0);
        int col = bn0 + wn + ni * 16 + (lane & 15);
        float av = aa[mi][ni][v];
        float gv = ag[mi][ni][v];
        float silu = gv / (1.f + __expf(-gv));
        Hout[(size_t)row * FFc + col] = (_Float16)(silu * av);
      }
}

// ---------- Flash-style cross attention, TDM-staged K/V tiles ----------
// grid: (Nn/128, Bb*HEADS), block 256 threads (8 waves, 16 query rows each).
// K chunk (64j x 64d, row stride DHc) and V^T chunk (64d x 64j, row stride Jj)
// are staged into LDS by TDM, double-buffered over the j loop, shared by all
// 8 waves. q fragments stay register-resident for the whole loop.
__global__ __launch_bounds__(256) void attention_kernel(
    const _Float16* __restrict__ Q,   // [b][i][h*64+d], scaled
    const _Float16* __restrict__ Kc,  // [b][j][d]
    const _Float16* __restrict__ VT,  // [b][d][j]
    _Float16* __restrict__ O) {       // [b][i][h*64+d]
  const int tid = threadIdx.x, lane = tid & 31, wave = tid >> 5;
  const int b = blockIdx.y / HEADS, h = blockIdx.y % HEADS;
  const int i0 = blockIdx.x * 128 + wave * 16;

  const _Float16* qb = Q  + (size_t)b * Nn * INNER + h * DHc;
  const _Float16* kb = Kc + (size_t)b * Jj * DHc;
  const _Float16* vb = VT + (size_t)b * DHc * Jj;

  const v16h q0 = load_am(qb, INNER, i0, 0,  lane);
  const v16h q1 = load_am(qb, INNER, i0, 32, lane);

  v8f o0 = vzero(), o1 = vzero(), o2 = vzero(), o3 = vzero();
  float mrow[8], lrow[8];
#pragma unroll
  for (int v = 0; v < 8; ++v) { mrow[v] = -1e30f; lrow[v] = 0.f; }

  __shared__ alignas(16) _Float16 Ks[2][64 * 64];   // [j_local][d]
  __shared__ alignas(16) _Float16 Vs[2][64 * 64];   // [d][j_local]
  __shared__ alignas(16) _Float16 P[8][16][64];

#if USE_TDM
  if (tid < 32) {
    tdm_load_tile_2d(kb, Ks[0], 64, DHc, Jj);
    tdm_load_tile_2d(vb, Vs[0], 64, Jj, DHc);
  }
#else
  stage_fallback(Ks[0], kb, 64, DHc, tid);
  stage_fallback(Vs[0], vb, 64, Jj, tid);
#endif

  for (int j0 = 0; j0 < Jj; j0 += 64) {
    const int cur = (j0 >> 6) & 1;
#if USE_TDM
    if (tid < 32) __builtin_amdgcn_s_wait_tensorcnt(0);
#endif
    __syncthreads();
    if (j0 + 64 < Jj) {
      const int nxt = cur ^ 1;
#if USE_TDM
      if (tid < 32) {
        tdm_load_tile_2d(kb + (size_t)(j0 + 64) * DHc, Ks[nxt], 64, DHc, Jj);
        tdm_load_tile_2d(vb + (j0 + 64),               Vs[nxt], 64, Jj,  DHc);
      }
#else
      stage_fallback(Ks[nxt], kb + (size_t)(j0 + 64) * DHc, 64, DHc, tid);
      stage_fallback(Vs[nxt], vb + (j0 + 64),               64, Jj,  tid);
#endif
    }
    const _Float16* Kl = Ks[cur];
    const _Float16* Vl = Vs[cur];

    v8f s0 = vzero(), s1 = vzero(), s2 = vzero(), s3 = vzero();
#pragma unroll
    for (int kk = 0; kk < 64; kk += 32) {
      v16h a = (kk == 0) ? q0 : q1;
      s0 = wmma_f16(a, load_bt(Kl, 64,  0, kk, lane), s0);
      s1 = wmma_f16(a, load_bt(Kl, 64, 16, kk, lane), s1);
      s2 = wmma_f16(a, load_bt(Kl, 64, 32, kk, lane), s2);
      s3 = wmma_f16(a, load_bt(Kl, 64, 48, kk, lane), s3);
    }
#pragma unroll
    for (int v = 0; v < 8; ++v) {
      float x0 = s0[v], x1 = s1[v], x2 = s2[v], x3 = s3[v];
      float mx = fmaxf(fmaxf(x0, x1), fmaxf(x2, x3));
#pragma unroll
      for (int off = 8; off >= 1; off >>= 1) mx = fmaxf(mx, __shfl_xor(mx, off, 32));
      float mnew = fmaxf(mrow[v], mx);
      float p0 = __expf(x0 - mnew), p1 = __expf(x1 - mnew);
      float p2 = __expf(x2 - mnew), p3 = __expf(x3 - mnew);
      float sum = p0 + p1 + p2 + p3;
#pragma unroll
      for (int off = 8; off >= 1; off >>= 1) sum += __shfl_xor(sum, off, 32);
      float alpha = __expf(mrow[v] - mnew);
      lrow[v] = lrow[v] * alpha + sum;
      mrow[v] = mnew;
      o0[v] *= alpha; o1[v] *= alpha; o2[v] *= alpha; o3[v] *= alpha;
      int r = v + ((lane & 16) ? 8 : 0);
      int c = lane & 15;
      P[wave][r][c +  0] = (_Float16)p0;
      P[wave][r][c + 16] = (_Float16)p1;
      P[wave][r][c + 32] = (_Float16)p2;
      P[wave][r][c + 48] = (_Float16)p3;
    }
    // P region is wave-private and LDS ops are in-order within a wave; the
    // fragment reloads below depend on the stores above via normal aliasing.
#pragma unroll
    for (int kk = 0; kk < 64; kk += 32) {
      v16h pa = load_am(&P[wave][0][0], 64, 0, kk, lane);
      o0 = wmma_f16(pa, load_bt(Vl, 64,  0, kk, lane), o0);
      o1 = wmma_f16(pa, load_bt(Vl, 64, 16, kk, lane), o1);
      o2 = wmma_f16(pa, load_bt(Vl, 64, 32, kk, lane), o2);
      o3 = wmma_f16(pa, load_bt(Vl, 64, 48, kk, lane), o3);
    }
  }
#pragma unroll
  for (int v = 0; v < 8; ++v) {
    float inv = 1.f / lrow[v];
    int row = i0 + v + ((lane & 16) ? 8 : 0);
    int c = lane & 15;
    size_t base = ((size_t)b * Nn + row) * INNER + h * DHc;
    O[base + c +  0] = (_Float16)(o0[v] * inv);
    O[base + c + 16] = (_Float16)(o1[v] * inv);
    O[base + c + 32] = (_Float16)(o2[v] * inv);
    O[base + c + 48] = (_Float16)(o3[v] * inv);
  }
}

// ---------- LayerNorm f32 -> f16 ----------
__global__ __launch_bounds__(256) void layernorm_f16(
    const float* __restrict__ X, const float* __restrict__ g,
    const float* __restrict__ bta, _Float16* __restrict__ out) {
  const int row = blockIdx.x;
  const float* xr = X + (size_t)row * DIMc;
  const int base = threadIdx.x * 4;
  float x4[4];
  float s = 0.f, ss = 0.f;
#pragma unroll
  for (int i = 0; i < 4; ++i) {
    float t = xr[base + i];
    x4[i] = t; s += t; ss += t * t;
  }
#pragma unroll
  for (int off = 16; off >= 1; off >>= 1) {
    s  += __shfl_xor(s,  off, 32);
    ss += __shfl_xor(ss, off, 32);
  }
  __shared__ float rs[8], rss[8];
  const int wave = threadIdx.x >> 5, lane = threadIdx.x & 31;
  if (lane == 0) { rs[wave] = s; rss[wave] = ss; }
  __syncthreads();
  s = 0.f; ss = 0.f;
#pragma unroll
  for (int w = 0; w < 8; ++w) { s += rs[w]; ss += rss[w]; }
  const float mu  = s * (1.f / DIMc);
  const float var = ss * (1.f / DIMc) - mu * mu;
  const float inv = rsqrtf(var + EPSc);
#pragma unroll
  for (int i = 0; i < 4; ++i) {
    int c = base + i;
    out[(size_t)row * DIMc + c] = (_Float16)((x4[i] - mu) * inv * g[c] + bta[c]);
  }
}

// ---------- weight convert+transpose: W f32 [K][N] -> WT f16 [N][K] ----------
__global__ void transpose_f32_to_f16(const float* __restrict__ W,
                                     _Float16* __restrict__ WT, int K, int N) {
  __shared__ float tile[32][33];
  const int nb = blockIdx.x * 32, kb = blockIdx.y * 32;
  for (int r = threadIdx.y; r < 32; r += 8)
    tile[r][threadIdx.x] = W[(size_t)(kb + r) * N + nb + threadIdx.x];
  __syncthreads();
  for (int r = threadIdx.y; r < 32; r += 8)
    WT[(size_t)(nb + r) * K + kb + threadIdx.x] = (_Float16)tile[threadIdx.x][r];
}

extern "C" void kernel_launch(void* const* d_in, const int* in_sizes, int n_in,
                              void* d_out, int out_size, void* d_ws, size_t ws_size,
                              hipStream_t stream) {
  (void)in_sizes; (void)n_in; (void)out_size; (void)ws_size;
  const float* x    = (const float*)d_in[0];
  const float* ctx  = (const float*)d_in[1];
  const float* ng   = (const float*)d_in[2];
  const float* nb   = (const float*)d_in[3];
  const float* cg   = (const float*)d_in[4];
  const float* cb   = (const float*)d_in[5];
  const float* Wq   = (const float*)d_in[6];
  const float* Wkv  = (const float*)d_in[7];
  const float* Wo   = (const float*)d_in[8];
  const float* Wff1 = (const float*)d_in[9];
  const float* Wff2 = (const float*)d_in[10];
  float* out = (float*)d_out;

  char* ws = (char*)d_ws;
  size_t off = 0;
  auto alloc = [&](size_t bytes) -> char* {
    char* p = ws + off;
    off += (bytes + 255) & ~(size_t)255;
    return p;
  };
  _Float16* xn    = (_Float16*)alloc((size_t)Bb * Nn * DIMc * 2);
  _Float16* cn    = (_Float16*)alloc((size_t)Bb * Jj * DIMc * 2);
  _Float16* WqT   = (_Float16*)alloc((size_t)INNER * DIMc * 2);
  _Float16* WkvT  = (_Float16*)alloc((size_t)2 * DHc * DIMc * 2);
  _Float16* WoT   = (_Float16*)alloc((size_t)DIMc * INNER * 2);
  _Float16* Wff1T = (_Float16*)alloc((size_t)2 * FFc * DIMc * 2);
  _Float16* Wff2T = (_Float16*)alloc((size_t)DIMc * FFc * 2);
  _Float16* qbuf  = (_Float16*)alloc((size_t)Bb * Nn * INNER * 2);
  _Float16* kbuf  = (_Float16*)alloc((size_t)Bb * Jj * DHc * 2);
  _Float16* vtbuf = (_Float16*)alloc((size_t)Bb * DHc * Jj * 2);
  _Float16* aout  = (_Float16*)alloc((size_t)Bb * Nn * INNER * 2);
  _Float16* hff   = (_Float16*)alloc((size_t)Bb * Nn * FFc * 2);

  const int M = Bb * Nn;  // 4096

  layernorm_f16<<<Bb * Nn, 256, 0, stream>>>(x,   ng, nb, xn);
  layernorm_f16<<<Bb * Jj, 256, 0, stream>>>(ctx, cg, cb, cn);

  transpose_f32_to_f16<<<dim3(INNER / 32, DIMc / 32), dim3(32, 8), 0, stream>>>(Wq, WqT, DIMc, INNER);
  transpose_f32_to_f16<<<dim3((2 * DHc) / 32, DIMc / 32), dim3(32, 8), 0, stream>>>(Wkv, WkvT, DIMc, 2 * DHc);
  transpose_f32_to_f16<<<dim3(DIMc / 32, INNER / 32), dim3(32, 8), 0, stream>>>(Wo, WoT, INNER, DIMc);
  transpose_f32_to_f16<<<dim3((2 * FFc) / 32, DIMc / 32), dim3(32, 8), 0, stream>>>(Wff1, Wff1T, DIMc, 2 * FFc);
  transpose_f32_to_f16<<<dim3(DIMc / 32, FFc / 32), dim3(32, 8), 0, stream>>>(Wff2, Wff2T, FFc, DIMc);

  gemm_f16out<<<dim3(M / 128, INNER / 128), 256, 0, stream>>>(xn, WqT, qbuf, DIMc, M, INNER, 0.125f);
  gemm_kv<<<dim3(M / 128, 1), 256, 0, stream>>>(cn, WkvT, kbuf, vtbuf, DIMc);
  attention_kernel<<<dim3(Nn / 128, Bb * HEADS), 256, 0, stream>>>(qbuf, kbuf, vtbuf, aout);
  gemm_f32store<<<dim3(M / 128, DIMc / 128), 256, 0, stream>>>(aout, WoT, out, INNER, M, DIMc);
  gemm_ff1_gated<<<dim3(M / 128, FFc / 64), 256, 0, stream>>>(xn, Wff1T, hff, DIMc);
  gemm_f32add<<<dim3(M / 128, DIMc / 128), 256, 0, stream>>>(hff, Wff2T, out, FFc, M, DIMc);
}
